// AttnBlock_77635828842587
// MI455X (gfx1250) — matrix operand order repaired and checked by reading, not compiled, not run on hardware
//
#include <hip/hip_runtime.h>
#include <hip/hip_bf16.h>

typedef __attribute__((ext_vector_type(16))) _Float16 v16h;
typedef __attribute__((ext_vector_type(8)))  _Float16 v8h;
typedef __attribute__((ext_vector_type(8)))  float    v8f;
typedef int gv4i __attribute__((vector_size(16)));   // matches builtin's V4i param

#define C_DIM 512
#define N_TOK 4096
#define NBATCH 4
#define ATTN_SCALE 0.04419417382415922f  // 512^-0.5

// --------------------------------------------------------------------------
// gfx1250 async Global->LDS path (ASYNCcnt). Guarded so the file compiles
// even if this toolchain lacks the builtins (falls back to sync LDS copy).
// Probe-learned signature: (int4 AS1* gsrc, int4 AS3* ldst, Ii offset, Ii cpol)
// --------------------------------------------------------------------------
#if defined(__has_builtin)
#  if __has_builtin(__builtin_amdgcn_global_load_async_to_lds_b128) && \
      __has_builtin(__builtin_amdgcn_s_wait_asynccnt)
#    define HAVE_ASYNC_LDS 1
#  endif
#endif
#ifndef HAVE_ASYNC_LDS
#  define HAVE_ASYNC_LDS 0
#endif

#if HAVE_ASYNC_LDS
#define ASYNC_CP_B128(gptr, lptr)                                         \
  __builtin_amdgcn_global_load_async_to_lds_b128(                         \
      (__attribute__((address_space(1))) gv4i*)(gptr),                    \
      (__attribute__((address_space(3))) gv4i*)(lptr), 0, 0)
#endif

// ---------------------------------------------------------------------------
// WMMA helper: D = A(16x32 f16) * B(32x16 f16) + C(16x16 f32)
// ---------------------------------------------------------------------------
__device__ __forceinline__ v8f wmma16(v16h a, v16h b, v8f c) {
  return __builtin_amdgcn_wmma_f32_16x16x32_f16(
      /*neg_a=*/false, a, /*neg_b=*/false, b,
      /*c_mod=*/(short)0, c, /*reuse_a=*/false, /*reuse_b=*/false);
}

// A fragment (16x32, row-major memory, f16). ISA layout:
// lane16 = row m; e in [0..7] -> K = half*8 + e ; e in [8..15] -> K = 16 + half*8 + e
__device__ __forceinline__ v16h load_a_f16(const _Float16* base, int row0, int ld,
                                           int k0, int lane) {
  const int l16 = lane & 15, hf = lane >> 4;
  const _Float16* p = base + (size_t)(row0 + l16) * ld + k0 + hf * 8;
  v8h lo = *(const v8h*)p;
  v8h hi = *(const v8h*)(p + 16);
  v16h r;
#pragma unroll
  for (int e = 0; e < 8; ++e) { r[e] = lo[e]; r[8 + e] = hi[e]; }
  return r;
}

// B fragment (32x16): column n = lane16; e in [0..7] -> K = half*8 + e ; e in [8..15] -> K = 16 + half*8 + e
// Memory layout: B(k,n) = mem[(col0+n)*ld + k0 + k]
__device__ __forceinline__ v16h load_b_f16(const _Float16* base, int col0, int ld,
                                           int k0, int lane) {
  const int l16 = lane & 15, hf = lane >> 4;
  const _Float16* p = base + (size_t)(col0 + l16) * ld + k0 + hf * 8;
  v8h lo = *(const v8h*)p;
  v8h hi = *(const v8h*)(p + 16);
  v16h r;
#pragma unroll
  for (int e = 0; e < 8; ++e) { r[e] = lo[e]; r[8 + e] = hi[e]; }
  return r;
}

// ---------------------------------------------------------------------------
// Weight convert: fp32 -> f16, run once per call (1 MB per matrix).
// ---------------------------------------------------------------------------
__global__ __launch_bounds__(256) void wcvt_kernel(const float* __restrict__ src,
                                                   _Float16* __restrict__ dst) {
  const int i = (blockIdx.x * 256 + threadIdx.x) * 4;
#pragma unroll
  for (int e = 0; e < 4; ++e) dst[i + e] = (_Float16)src[i + e];
}

// ---------------------------------------------------------------------------
// GroupNorm: one block per (batch, group). Writes f16 token-major [n, c].
// ---------------------------------------------------------------------------
__global__ __launch_bounds__(256) void groupnorm_kernel(
    const float* __restrict__ x, const float* __restrict__ gscale,
    const float* __restrict__ gbias, _Float16* __restrict__ hn) {
  const int bg = blockIdx.x;
  const int b = bg >> 5, g = bg & 31;
  const int lane = threadIdx.x & 31, wave = threadIdx.x >> 5;
  const float* xb = x + ((size_t)b * C_DIM + g * 16) * N_TOK;

  float s = 0.f, ss = 0.f;
  for (int i = threadIdx.x; i < 16 * N_TOK; i += 256) {
    float v = xb[i];
    s += v; ss += v * v;
  }
#pragma unroll
  for (int m = 1; m < 32; m <<= 1) {
    s  += __shfl_xor(s, m, 32);
    ss += __shfl_xor(ss, m, 32);
  }
  __shared__ float rs[8], rss[8], stat[2];
  if (lane == 0) { rs[wave] = s; rss[wave] = ss; }
  __syncthreads();
  if (threadIdx.x == 0) {
    float S = 0.f, SS = 0.f;
#pragma unroll
    for (int w = 0; w < 8; ++w) { S += rs[w]; SS += rss[w]; }
    const float inv_n = 1.0f / (16.0f * N_TOK);
    float mean = S * inv_n;
    float var = SS * inv_n - mean * mean;
    stat[0] = mean;
    stat[1] = rsqrtf(var + 1e-5f);
  }
  __syncthreads();
  const float mean = stat[0], rstd = stat[1];
  _Float16* hb = hn + (size_t)b * N_TOK * C_DIM;
  for (int i = threadIdx.x; i < 16 * N_TOK; i += 256) {
    int cl = i >> 12;
    int n  = i & (N_TOK - 1);
    int c  = g * 16 + cl;
    float v = (xb[i] - mean) * rstd * gscale[c] + gbias[c];
    hb[(size_t)n * C_DIM + c] = (_Float16)v;
  }
}

// ---------------------------------------------------------------------------
// GEMM: Y[o,n] = sum_c Wh[o,c] * Bt(c,n) + bias[o]
//  Wh f16 [512x512] row-major (preconverted); Bt token-major f16.
//  A-tile (64x32, shared by 4 waves) staged in LDS via double-buffered async
//  Global->LDS copy. 8 waves/block, each computes a 32x32 tile.
//  mode 0: f16 channel-major; mode 1: f16 token-major; mode 2: f32 + residual
// ---------------------------------------------------------------------------
__global__ __launch_bounds__(256) void gemm512_kernel(
    const _Float16* __restrict__ Wh, const float* __restrict__ bias,
    const _Float16* __restrict__ Bt, _Float16* __restrict__ out16,
    float* __restrict__ outf, const float* __restrict__ xres, int mode) {
  const int b = blockIdx.z;
  const int lane = threadIdx.x & 31;
  const int wave = threadIdx.x >> 5;
  const int wm = wave >> 2, wn = wave & 3;
  const int mblk = blockIdx.x * 64;
  const int m0 = mblk + wm * 32;
  const int n0 = blockIdx.y * 128 + wn * 32;
  const _Float16* Bb = Bt + (size_t)b * N_TOK * C_DIM;
  const int l16 = lane & 15, hf = lane >> 4;

  __shared__ _Float16 As[2][64][72];     // padded stride: 144 B/row, no bank conflicts
  const int crow = threadIdx.x >> 2;     // 0..63 : row of the A tile this thread copies
  const int cseg = (threadIdx.x & 3) * 8;

  v8f acc[2][2] = {};

  // issue copy of k-tile into buffer
  auto issue_copy = [&](int k0, int buf) {
    const _Float16* g = Wh + (size_t)(mblk + crow) * C_DIM + k0 + cseg;
    _Float16* l = &As[buf][crow][cseg];
#if HAVE_ASYNC_LDS
    ASYNC_CP_B128(g, l);
#else
    *(v8h*)l = *(const v8h*)g;
#endif
  };

  issue_copy(0, 0);
  for (int kt = 0; kt < 16; ++kt) {
    const int buf = kt & 1;
    const int k0 = kt * 32;
    if (kt + 1 < 16) {
      issue_copy(k0 + 32, buf ^ 1);
#if HAVE_ASYNC_LDS
      __builtin_amdgcn_s_wait_asynccnt(1);   // previous tile landed; newest in flight
#endif
    } else {
#if HAVE_ASYNC_LDS
      __builtin_amdgcn_s_wait_asynccnt(0);
#endif
    }
    __syncthreads();

    if (kt + 2 < 16)  // prefetch the B stream two tiles ahead -> global_prefetch_b8
      __builtin_prefetch(Bb + (size_t)(n0 + l16) * C_DIM + k0 + 64, 0, 0);

    v16h a0 = load_a_f16(&As[buf][0][0], wm * 32,      72, 0, lane);
    v16h a1 = load_a_f16(&As[buf][0][0], wm * 32 + 16, 72, 0, lane);
    v16h b0 = load_b_f16(Bb, n0,      C_DIM, k0, lane);
    v16h b1 = load_b_f16(Bb, n0 + 16, C_DIM, k0, lane);
    acc[0][0] = wmma16(a0, b0, acc[0][0]);
    acc[0][1] = wmma16(a0, b1, acc[0][1]);
    acc[1][0] = wmma16(a1, b0, acc[1][0]);
    acc[1][1] = wmma16(a1, b1, acc[1][1]);
    __syncthreads();   // all waves done with buf before it is overwritten
  }

#pragma unroll
  for (int mi = 0; mi < 2; ++mi)
#pragma unroll
    for (int ni = 0; ni < 2; ++ni)
#pragma unroll
      for (int r = 0; r < 8; ++r) {
        int m = m0 + mi * 16 + hf * 8 + r;
        int n = n0 + ni * 16 + l16;
        float v = acc[mi][ni][r] + bias[m];
        if (mode == 0) {
          out16[((size_t)b * C_DIM + m) * N_TOK + n] = (_Float16)v;
        } else if (mode == 1) {
          out16[((size_t)b * N_TOK + n) * C_DIM + m] = (_Float16)v;
        } else {
          size_t idx = ((size_t)b * C_DIM + m) * N_TOK + n;
          outf[idx] = v + xres[idx];
        }
      }
}

// ---------------------------------------------------------------------------
// Attention pass A: per-row softmax stats for S = (Q K^T)*scale.
// ---------------------------------------------------------------------------
__global__ __launch_bounds__(256) void attn_stats_kernel(
    const _Float16* __restrict__ q, const _Float16* __restrict__ k,
    float* __restrict__ ms, float* __restrict__ ls) {
  const int b = blockIdx.y;
  const int i0 = blockIdx.x * 16;
  const int lane = threadIdx.x & 31, wave = threadIdx.x >> 5;
  const _Float16* qb = q + (size_t)b * N_TOK * C_DIM;
  const _Float16* kb = k + (size_t)b * N_TOK * C_DIM;

  v16h aq[16];
#pragma unroll
  for (int kk = 0; kk < 16; ++kk)
    aq[kk] = load_a_f16(qb, i0, C_DIM, kk * 32, lane);

  float rmax[8], rsum[8];
#pragma unroll
  for (int r = 0; r < 8; ++r) { rmax[r] = -1e30f; rsum[r] = 0.f; }

  for (int jt = wave; jt < N_TOK / 16; jt += 8) {
    const int j0 = jt * 16;
    v8f s = {};
#pragma unroll
    for (int kk = 0; kk < 16; ++kk) {
      v16h bk = load_b_f16(kb, j0, C_DIM, kk * 32, lane);
      s = wmma16(aq[kk], bk, s);
    }
#pragma unroll
    for (int r = 0; r < 8; ++r) {
      float v = s[r] * ATTN_SCALE;
      float tmax = v;
#pragma unroll
      for (int m = 1; m < 16; m <<= 1)
        tmax = fmaxf(tmax, __shfl_xor(tmax, m, 32));
      float e = __expf(v - tmax);
#pragma unroll
      for (int m = 1; m < 16; m <<= 1)
        e += __shfl_xor(e, m, 32);
      float nm = fmaxf(rmax[r], tmax);
      rsum[r] = rsum[r] * __expf(rmax[r] - nm) + e * __expf(tmax - nm);
      rmax[r] = nm;
    }
  }

  __shared__ float sm[8][16], sl[8][16];
  const int l16 = lane & 15, hf = lane >> 4;
  if (l16 == 0) {
#pragma unroll
    for (int r = 0; r < 8; ++r) {
      sm[wave][hf * 8 + r] = rmax[r];
      sl[wave][hf * 8 + r] = rsum[r];
    }
  }
  __syncthreads();
  if (threadIdx.x < 16) {
    float gm = -1e30f, gl = 0.f;
#pragma unroll
    for (int w = 0; w < 8; ++w) gm = fmaxf(gm, sm[w][threadIdx.x]);
#pragma unroll
    for (int w = 0; w < 8; ++w) gl += sl[w][threadIdx.x] * __expf(sm[w][threadIdx.x] - gm);
    ms[(size_t)b * N_TOK + i0 + threadIdx.x] = gm;
    ls[(size_t)b * N_TOK + i0 + threadIdx.x] = gl;
  }
}

// ---------------------------------------------------------------------------
// Attention pass B: O = softmax(S) V using precomputed stats. P staged in LDS.
// ---------------------------------------------------------------------------
__global__ __launch_bounds__(256) void attn_out_kernel(
    const _Float16* __restrict__ q, const _Float16* __restrict__ k,
    const _Float16* __restrict__ v, const float* __restrict__ ms,
    const float* __restrict__ ls, _Float16* __restrict__ o) {
  const int b = blockIdx.y;
  const int i0 = blockIdx.x * 32;
  const int lane = threadIdx.x & 31, wave = threadIdx.x >> 5;
  const int l16 = lane & 15, hf = lane >> 4;
  const int wm = wave >> 2;
  const int wn = wave & 3;
  const _Float16* qb = q + (size_t)b * N_TOK * C_DIM;
  const _Float16* kb = k + (size_t)b * N_TOK * C_DIM;
  const _Float16* vb = v + (size_t)b * C_DIM * N_TOK;  // channel-major

  v16h aq[16];
#pragma unroll
  for (int kk = 0; kk < 16; ++kk)
    aq[kk] = load_a_f16(qb, i0 + wm * 16, C_DIM, kk * 32, lane);

  float mrow[8], invl[8];
#pragma unroll
  for (int r = 0; r < 8; ++r) {
    int row = i0 + wm * 16 + hf * 8 + r;
    mrow[r] = ms[(size_t)b * N_TOK + row];
    invl[r] = 1.0f / ls[(size_t)b * N_TOK + row];
  }

  __shared__ _Float16 P[32][72];
  v8f acc[2][4] = {};

  for (int j0 = 0; j0 < N_TOK; j0 += 64) {
    v8f s = {};
#pragma unroll
    for (int kk = 0; kk < 16; ++kk) {
      v16h bk = load_b_f16(kb, j0 + wn * 16, C_DIM, kk * 32, lane);
      s = wmma16(aq[kk], bk, s);
    }
#pragma unroll
    for (int r = 0; r < 8; ++r) {
      float p = __expf(s[r] * ATTN_SCALE - mrow[r]) * invl[r];
      P[wm * 16 + hf * 8 + r][wn * 16 + l16] = (_Float16)p;
    }
    __syncthreads();

    const int c0 = wave * 64;
#pragma unroll
    for (int kt = 0; kt < 2; ++kt) {
      v16h pa0 = load_a_f16(&P[0][0], 0,  72, kt * 32, lane);
      v16h pa1 = load_a_f16(&P[0][0], 16, 72, kt * 32, lane);
#pragma unroll
      for (int nt = 0; nt < 4; ++nt) {
        v16h bv = load_b_f16(vb, c0 + nt * 16, N_TOK, j0 + kt * 32, lane);
        acc[0][nt] = wmma16(pa0, bv, acc[0][nt]);
        acc[1][nt] = wmma16(pa1, bv, acc[1][nt]);
      }
    }
    __syncthreads();
  }

  _Float16* ob = o + (size_t)b * N_TOK * C_DIM;
#pragma unroll
  for (int mi = 0; mi < 2; ++mi)
#pragma unroll
    for (int nt = 0; nt < 4; ++nt)
#pragma unroll
      for (int r = 0; r < 8; ++r) {
        int row = i0 + mi * 16 + hf * 8 + r;
        int c   = wave * 64 + nt * 16 + l16;
        ob[(size_t)row * C_DIM + c] = (_Float16)acc[mi][nt][r];
      }
}

// ---------------------------------------------------------------------------
extern "C" void kernel_launch(void* const* d_in, const int* in_sizes, int n_in,
                              void* d_out, int out_size, void* d_ws, size_t ws_size,
                              hipStream_t stream) {
  const float* x      = (const float*)d_in[0];
  const float* nscale = (const float*)d_in[1];
  const float* nbias  = (const float*)d_in[2];
  const float* wq = (const float*)d_in[3];
  const float* bq = (const float*)d_in[4];
  const float* wk = (const float*)d_in[5];
  const float* bk = (const float*)d_in[6];
  const float* wv = (const float*)d_in[7];
  const float* bv = (const float*)d_in[8];
  const float* wo = (const float*)d_in[9];
  const float* bo = (const float*)d_in[10];
  float* out = (float*)d_out;

  const size_t SZ = (size_t)NBATCH * N_TOK * C_DIM;  // 8M f16 elements / buffer
  const size_t WSZ = (size_t)C_DIM * C_DIM;          // 256K f16 / weight matrix
  _Float16* hn = (_Float16*)d_ws;     // token-major
  _Float16* qm = hn + SZ;             // token-major
  _Float16* km = qm + SZ;             // token-major
  _Float16* vm = km + SZ;             // channel-major
  _Float16* om = vm + SZ;             // token-major
  float* ms = (float*)(om + SZ);
  float* ls = ms + (size_t)NBATCH * N_TOK;
  _Float16* wqh = (_Float16*)(ls + (size_t)NBATCH * N_TOK);
  _Float16* wkh = wqh + WSZ;
  _Float16* wvh = wkh + WSZ;
  _Float16* woh = wvh + WSZ;

  // 0) convert weights fp32 -> f16 (once per call)
  const int wgrid = (int)(WSZ / (256 * 4));
  wcvt_kernel<<<wgrid, 256, 0, stream>>>(wq, wqh);
  wcvt_kernel<<<wgrid, 256, 0, stream>>>(wk, wkh);
  wcvt_kernel<<<wgrid, 256, 0, stream>>>(wv, wvh);
  wcvt_kernel<<<wgrid, 256, 0, stream>>>(wo, woh);

  // 1) GroupNorm -> hn (token-major f16)
  groupnorm_kernel<<<NBATCH * 32, 256, 0, stream>>>(x, nscale, nbias, hn);

  // 2) q/k/v projections (WMMA GEMMs, async-LDS staged A tiles)
  dim3 gg(C_DIM / 64, N_TOK / 128, NBATCH);
  gemm512_kernel<<<gg, 256, 0, stream>>>(wqh, bq, hn, qm, nullptr, nullptr, 1);
  gemm512_kernel<<<gg, 256, 0, stream>>>(wkh, bk, hn, km, nullptr, nullptr, 1);
  gemm512_kernel<<<gg, 256, 0, stream>>>(wvh, bv, hn, vm, nullptr, nullptr, 0);

  // 3) softmax row stats
  attn_stats_kernel<<<dim3(N_TOK / 16, NBATCH), 256, 0, stream>>>(qm, km, ms, ls);

  // 4) attention output O = softmax(S) V
  attn_out_kernel<<<dim3(N_TOK / 32, NBATCH), 256, 0, stream>>>(qm, km, vm, ms, ls, om);

  // 5) output projection + bias + residual (fp32 out)
  gemm512_kernel<<<gg, 256, 0, stream>>>(woh, bo, om, nullptr, out, x, 2);
}
